// DualSQL_22565758173342
// MI455X (gfx1250) — matrix operand order repaired and checked by reading, compiled, not run on hardware
//
#include <hip/hip_runtime.h>

// ---------------- model constants ----------------
#define EMBD   300
#define HSZ    150
#define DECSZ  300
#define VU_    50000
#define VQ_    30000
#define VS_    5000
#define NCOLS  64
#define COLLEN 6
#define NTURNS 6
#define INLEN  128
#define GENLEN 64
#define NOUT   (VQ_ + NCOLS)      // 30064
#define STEPS  (GENLEN - 1)       // 63

typedef __attribute__((ext_vector_type(16))) _Float16 v16h;
typedef __attribute__((ext_vector_type(8)))  float    v8f;

__device__ __forceinline__ float sigf(float x) { return 1.0f / (1.0f + expf(-x)); }
__host__ __device__ __forceinline__ int clampi(int x, int lo, int hi) {
  return x < lo ? lo : (x > hi ? hi : x);
}
__host__ __device__ __forceinline__ int imin2(int a, int b) { return a < b ? a : b; }

// ===================================================================
// WMMA fragment loaders (f32 -> f16 in-register), ISA 7.12.2 layouts
// ===================================================================
__device__ __forceinline__ v16h load_afrag(const float* __restrict__ A, int lda,
                                           int m, int M, int k0, int K, int half) {
  v16h a;
  #pragma unroll
  for (int i = 0; i < 8; ++i) {
    int kb = ((i < 4) ? (2 * i) : (16 + 2 * (i - 4))) + half * 8;
    int k  = k0 + kb;
    float a0 = (m < M && (k    ) < K) ? A[(size_t)m * lda + k    ] : 0.0f;
    float a1 = (m < M && (k + 1) < K) ? A[(size_t)m * lda + k + 1] : 0.0f;
    a[2 * i]     = (_Float16)a0;
    a[2 * i + 1] = (_Float16)a1;
  }
  return a;
}

__device__ __forceinline__ v16h load_bfrag(const float* __restrict__ W, int ldw, int bT,
                                           int n, int N, int k0, int K, int half) {
  v16h b;
  #pragma unroll
  for (int i = 0; i < 8; ++i) {
    int k = k0 + ((i < 4) ? (2 * i) : (16 + 2 * (i - 4))) + half * 8;
    float b0 = 0.0f, b1 = 0.0f;
    if (n < N) {
      if (bT) {
        b0 = ((k    ) < K) ? W[(size_t)n * ldw + k    ] : 0.0f;
        b1 = ((k + 1) < K) ? W[(size_t)n * ldw + k + 1] : 0.0f;
      } else {
        b0 = ((k    ) < K) ? W[(size_t)(k    ) * ldw + n] : 0.0f;
        b1 = ((k + 1) < K) ? W[(size_t)(k + 1) * ldw + n] : 0.0f;
      }
    }
    b[2 * i]     = (_Float16)b0;
    b[2 * i + 1] = (_Float16)b1;
  }
  return b;
}

// ===================================================================
// Generic WMMA GEMM:  C[m][n0+n] = beta*C + sum_k A[m][k]*B[k][n] (+bias[n])
//   bT=1: W is [N,K] row-major (C = A * W^T);  bT=0: W is [K,N] row-major.
// One wave32 computes a 16x64 output strip (4 N-tiles): the A fragment is
// loaded once per K-block and reused by 4 back-to-back independent WMMAs.
// ===================================================================
__global__ __launch_bounds__(32)
void wmma_gemm_f16(const float* __restrict__ A, int lda,
                   const float* __restrict__ W, int ldw, int bT,
                   const float* __restrict__ bias, float beta,
                   float* __restrict__ C, int ldc, int n0,
                   int M, int N, int K) {
  const int lane = threadIdx.x;            // 0..31
  const int half = lane >> 4;              // 0/1
  const int l16  = lane & 15;
  const int mbase  = blockIdx.y * 16;
  const int nbase0 = blockIdx.x * 64;
  const int m = mbase + l16;
  v8f acc[4] = {v8f{}, v8f{}, v8f{}, v8f{}};

  #pragma unroll 2
  for (int k0 = 0; k0 < K; k0 += 32) {
    if (k0 + 32 < K) {
      // warm next K-block of the (large) W operand through the cache hierarchy
      int np = nbase0 + l16;
      __builtin_prefetch((const void*)(bT ? (W + (size_t)(np < N ? np : 0) * ldw + k0 + 32)
                                          : (W + (size_t)(k0 + 32) * ldw + (np < N ? np : 0))),
                         0, 1);
    }
    v16h a = load_afrag(A, lda, m, M, k0, K, half);
    v16h b0 = load_bfrag(W, ldw, bT, nbase0      + l16, N, k0, K, half);
    acc[0] = __builtin_amdgcn_wmma_f32_16x16x32_f16(false, a, false, b0, (short)0, acc[0],
                                                    false, false);
    v16h b1 = load_bfrag(W, ldw, bT, nbase0 + 16 + l16, N, k0, K, half);
    acc[1] = __builtin_amdgcn_wmma_f32_16x16x32_f16(false, a, false, b1, (short)0, acc[1],
                                                    false, false);
    v16h b2 = load_bfrag(W, ldw, bT, nbase0 + 32 + l16, N, k0, K, half);
    acc[2] = __builtin_amdgcn_wmma_f32_16x16x32_f16(false, a, false, b2, (short)0, acc[2],
                                                    false, false);
    v16h b3 = load_bfrag(W, ldw, bT, nbase0 + 48 + l16, N, k0, K, half);
    acc[3] = __builtin_amdgcn_wmma_f32_16x16x32_f16(false, a, false, b3, (short)0, acc[3],
                                                    false, false);
  }

  // C/D layout: VGPR r holds M = r + 8*half, N = l16 (per 16x16 tile)
  #pragma unroll
  for (int s = 0; s < 4; ++s) {
    #pragma unroll
    for (int r = 0; r < 8; ++r) {
      int mm = mbase + half * 8 + r;
      int nn = nbase0 + s * 16 + l16;
      if (mm < M && nn < N) {
        float v = acc[s][r];
        if (bias) v += bias[nn];
        float* p = C + (size_t)mm * ldc + n0 + nn;
        *p = (beta != 0.0f ? beta * (*p) : 0.0f) + v;
      }
    }
  }
}

static void gemm(const float* A, int lda, const float* W, int ldw, int bT,
                 const float* bias, float beta, float* C, int ldc, int n0,
                 int M, int N, int K, hipStream_t stream) {
  dim3 g((N + 63) / 64, (M + 15) / 16);
  wmma_gemm_f16<<<g, 32, 0, stream>>>(A, lda, W, ldw, bT, bias, beta, C, ldc, n0, M, N, K);
}

// ===================================================================
// Embedding gathers
// ===================================================================
__global__ void k_embed_input(const int* __restrict__ seq,
                              const float* __restrict__ utter,
                              const float* __restrict__ S,
                              float* __restrict__ x) {
  int t = blockIdx.x;
  int idx = seq[t];
  const float* src = (idx < VU_) ? (utter + (size_t)clampi(idx, 0, VU_ - 1) * EMBD)
                                 : (S + (size_t)clampi(idx - VU_, 0, NCOLS - 1) * EMBD);
  for (int d = threadIdx.x; d < EMBD; d += blockDim.x) x[(size_t)t * EMBD + d] = src[d];
}

__global__ void k_coltok_embed(const int* __restrict__ toks,
                               const float* __restrict__ emb,
                               float* __restrict__ x) {
  int row = blockIdx.x;                       // col*6 + l
  int tok = clampi(toks[row], 0, VS_ - 1);
  const float* src = emb + (size_t)tok * EMBD;
  for (int d = threadIdx.x; d < EMBD; d += blockDim.x) x[(size_t)row * EMBD + d] = src[d];
}

// ===================================================================
// Sequential LSTM scans (recurrent matvec only; input proj precomputed by WMMA)
// ===================================================================
__global__ void k_lstm_scan(const float* __restrict__ zin, int steps,
                            const float* __restrict__ Whh_base,
                            float* __restrict__ yout,
                            float* __restrict__ hfin, float* __restrict__ cfin) {
  int dir = blockIdx.x;
  const float* Whh = Whh_base + (size_t)dir * 4 * HSZ * HSZ;   // [600,150]
  __shared__ float h[HSZ], c[HSZ], z[4 * HSZ];
  int tid = threadIdx.x, nt = blockDim.x;
  for (int d = tid; d < HSZ; d += nt) { h[d] = 0.0f; c[d] = 0.0f; }
  __syncthreads();
  for (int s = 0; s < steps; ++s) {
    int t = dir ? (steps - 1 - s) : s;
    for (int r = tid; r < 4 * HSZ; r += nt) {
      float acc = zin[(size_t)t * (8 * HSZ) + dir * 4 * HSZ + r];
      const float* w = Whh + (size_t)r * HSZ;
      for (int j = 0; j < HSZ; ++j) acc += w[j] * h[j];
      z[r] = acc;
    }
    __syncthreads();
    for (int d = tid; d < HSZ; d += nt) {
      float ig = sigf(z[d]), fg = sigf(z[HSZ + d]);
      float gg = tanhf(z[2 * HSZ + d]), og = sigf(z[3 * HSZ + d]);
      float cn = fg * c[d] + ig * gg;
      float hn = og * tanhf(cn);
      c[d] = cn; h[d] = hn;
      if (yout) yout[(size_t)t * EMBD + dir * HSZ + d] = hn;
    }
    __syncthreads();
  }
  for (int d = tid; d < HSZ; d += nt) { hfin[dir * HSZ + d] = h[d]; cfin[dir * HSZ + d] = c[d]; }
}

__global__ void k_schema_scan(const float* __restrict__ zin,
                              const float* __restrict__ Whh_base,
                              float* __restrict__ S) {
  int col = blockIdx.x >> 1, dir = blockIdx.x & 1;
  const float* Whh = Whh_base + (size_t)dir * 4 * HSZ * HSZ;
  __shared__ float h[HSZ], c[HSZ], z[4 * HSZ];
  int tid = threadIdx.x, nt = blockDim.x;
  for (int d = tid; d < HSZ; d += nt) { h[d] = 0.0f; c[d] = 0.0f; }
  __syncthreads();
  for (int s = 0; s < COLLEN; ++s) {
    int t = dir ? (COLLEN - 1 - s) : s;
    for (int r = tid; r < 4 * HSZ; r += nt) {
      float acc = zin[(size_t)(col * COLLEN + t) * (8 * HSZ) + dir * 4 * HSZ + r];
      const float* w = Whh + (size_t)r * HSZ;
      for (int j = 0; j < HSZ; ++j) acc += w[j] * h[j];
      z[r] = acc;
    }
    __syncthreads();
    for (int d = tid; d < HSZ; d += nt) {
      float ig = sigf(z[d]), fg = sigf(z[HSZ + d]);
      float gg = tanhf(z[2 * HSZ + d]), og = sigf(z[3 * HSZ + d]);
      float cn = fg * c[d] + ig * gg;
      c[d] = cn; h[d] = og * tanhf(cn);
    }
    __syncthreads();
  }
  for (int d = tid; d < HSZ; d += nt) S[(size_t)col * EMBD + dir * HSZ + d] = h[d];
}

__global__ void k_save_state(const float* __restrict__ fh, const float* __restrict__ fc,
                             float* __restrict__ ph, float* __restrict__ pc) {
  int d = threadIdx.x;
  if (d < EMBD) { ph[d] = fh[d]; pc[d] = fc[d]; }
}

// ===================================================================
// Row softmax / log-softmax / argmax-embed
// ===================================================================
__global__ void k_softmax_rows(float* __restrict__ A, int n) {
  float* row = A + (size_t)blockIdx.x * n;
  __shared__ float red[128];
  int tid = threadIdx.x;
  float mx = -1e30f;
  for (int j = tid; j < n; j += 128) mx = fmaxf(mx, row[j]);
  red[tid] = mx; __syncthreads();
  if (tid == 0) { float m = -1e30f; for (int k = 0; k < 128; ++k) m = fmaxf(m, red[k]); red[0] = m; }
  __syncthreads();
  mx = red[0]; __syncthreads();
  float s = 0.0f;
  for (int j = tid; j < n; j += 128) { float e = expf(row[j] - mx); row[j] = e; s += e; }
  red[tid] = s; __syncthreads();
  if (tid == 0) { float t = 0.0f; for (int k = 0; k < 128; ++k) t += red[k]; red[0] = t; }
  __syncthreads();
  float inv = 1.0f / red[0];
  for (int j = tid; j < n; j += 128) row[j] *= inv;
}

__global__ void k_logsoftmax_rows(float* __restrict__ A, int n) {
  float* row = A + (size_t)blockIdx.x * n;
  __shared__ float red[256];
  int tid = threadIdx.x;
  float mx = -1e30f;
  for (int j = tid; j < n; j += 256) mx = fmaxf(mx, row[j]);
  red[tid] = mx; __syncthreads();
  if (tid == 0) { float m = -1e30f; for (int k = 0; k < 256; ++k) m = fmaxf(m, red[k]); red[0] = m; }
  __syncthreads();
  mx = red[0]; __syncthreads();
  float s = 0.0f;
  for (int j = tid; j < n; j += 256) s += expf(row[j] - mx);
  red[tid] = s; __syncthreads();
  if (tid == 0) { float t = 0.0f; for (int k = 0; k < 256; ++k) t += red[k]; red[0] = logf(t); }
  __syncthreads();
  float lse = mx + red[0];
  for (int j = tid; j < n; j += 256) row[j] -= lse;
}

__global__ void k_amemb(const float* __restrict__ dists,
                        const float* __restrict__ query_emb,
                        const float* __restrict__ S,
                        float* __restrict__ slot) {
  int t = blockIdx.x;
  const float* row = dists + (size_t)t * NOUT;
  __shared__ float bv[256]; __shared__ int bi[256]; __shared__ int tok;
  int tid = threadIdx.x;
  float mv = -1e30f; int mi = 0;
  for (int j = tid; j < NOUT; j += 256)
    if (row[j] > mv) { mv = row[j]; mi = j; }
  bv[tid] = mv; bi[tid] = mi; __syncthreads();
  if (tid == 0) {
    float best = -1e30f; int besti = 0;
    for (int k = 0; k < 256; ++k)
      if (bv[k] > best || (bv[k] == best && bi[k] < besti)) { best = bv[k]; besti = bi[k]; }
    tok = besti;
  }
  __syncthreads();
  int tk = tok;
  const float* src = (tk < VQ_) ? (query_emb + (size_t)clampi(tk, 0, VQ_ - 1) * EMBD)
                                : (S + (size_t)clampi(tk - VQ_, 0, NCOLS - 1) * EMBD);
  for (int d = tid; d < EMBD; d += 256) slot[(size_t)(t + 1) * EMBD + d] = src[d];
  if (t == 0)  // BOS row = query_emb[0]
    for (int d = tid; d < EMBD; d += 256) slot[d] = query_emb[d];
}

// ===================================================================
// Decoder: sequential 63-step scan in one workgroup (32 waves)
// ===================================================================
__device__ void attn_ctx(int tid, int nt, const float* q, const float* Wq,
                         const float* vv, const float* keys, const float* KW, int nk,
                         float* qW, float* sc, float* red, float* ctx) {
  for (int d = tid; d < EMBD; d += nt) {
    float s = 0.0f;
    for (int k = 0; k < EMBD; ++k) s += q[k] * Wq[(size_t)k * EMBD + d];
    qW[d] = s;
  }
  __syncthreads();
  for (int j = tid; j < nk; j += nt) {
    const float* kw = KW + (size_t)j * EMBD;
    float s = 0.0f;
    for (int d = 0; d < EMBD; ++d) s += vv[d] * tanhf(qW[d] + kw[d]);
    sc[j] = s;
  }
  __syncthreads();
  if (tid < 64) {
    float mx = -1e30f;
    for (int j = tid; j < nk; j += 64) mx = fmaxf(mx, sc[j]);
    red[tid] = mx;
  }
  __syncthreads();
  if (tid == 0) { float m = -1e30f; for (int k = 0; k < 64; ++k) m = fmaxf(m, red[k]); red[0] = m; }
  __syncthreads();
  float mx = red[0];
  __syncthreads();
  if (tid < 64) {
    float s = 0.0f;
    for (int j = tid; j < nk; j += 64) { float e = expf(sc[j] - mx); sc[j] = e; s += e; }
    red[tid] = s;
  }
  __syncthreads();
  if (tid == 0) { float s = 0.0f; for (int k = 0; k < 64; ++k) s += red[k]; red[0] = s; }
  __syncthreads();
  float inv = 1.0f / red[0];
  for (int d = tid; d < EMBD; d += nt) {
    float s = 0.0f;
    for (int j = 0; j < nk; ++j) s += sc[j] * keys[(size_t)j * EMBD + d];
    ctx[d] = s * inv;
  }
  __syncthreads();
}

__global__ __launch_bounds__(1024)
void k_decoder(const float* __restrict__ fh, const float* __restrict__ fc,
               const float* __restrict__ prev_h, const float* __restrict__ prev_c, int nh,
               const float* __restrict__ S, const float* __restrict__ KWs,
               const float* __restrict__ keys_in, const float* __restrict__ KWi, int nki,
               const float* __restrict__ keys_out, const float* __restrict__ KWo, int nko,
               const float* __restrict__ Wq, const float* __restrict__ Wk,
               const float* __restrict__ vA,
               const float* __restrict__ Wih0, const float* __restrict__ Whh0,
               const float* __restrict__ b0,
               const float* __restrict__ Wih1, const float* __restrict__ Whh1,
               const float* __restrict__ b1,
               const float* __restrict__ Wt,
               const int* __restrict__ gt_row, const float* __restrict__ query_emb,
               float* __restrict__ out_vecs) {
  __shared__ float dh[EMBD], h1[EMBD], c1[EMBD], h2[EMBD], c2[EMBD];
  __shared__ float xin[4 * EMBD];   // [out_emb | ctx_s | ctx_i | ctx_o]
  __shared__ float z[4 * DECSZ];
  __shared__ float qW[EMBD], sc[640], red[64], outv[EMBD], kwh[5 * EMBD];
  int tid = threadIdx.x, nt = blockDim.x;

  // ---- state init (history attention over prior turns) ----
  for (int d = tid; d < EMBD; d += nt) { dh[d] = fh[d]; c1[d] = fc[d]; }
  __syncthreads();
  if (nh > 0) {
    for (int idx = tid; idx < nh * EMBD; idx += nt) {
      int j = idx / EMBD, d = idx % EMBD;
      const float* hk = prev_h + (size_t)j * EMBD;
      float s = 0.0f;
      for (int k = 0; k < EMBD; ++k) s += hk[k] * Wk[(size_t)k * EMBD + d];
      kwh[idx] = s;
    }
    __syncthreads();
    attn_ctx(tid, nt, dh, Wq, vA, prev_h, kwh, nh, qW, sc, red, outv);
    for (int d = tid; d < EMBD; d += nt) dh[d] += outv[d];     // h0n
    __syncthreads();
    for (int idx = tid; idx < nh * EMBD; idx += nt) {
      int j = idx / EMBD, d = idx % EMBD;
      const float* hk = prev_c + (size_t)j * EMBD;
      float s = 0.0f;
      for (int k = 0; k < EMBD; ++k) s += hk[k] * Wk[(size_t)k * EMBD + d];
      kwh[idx] = s;
    }
    __syncthreads();
    attn_ctx(tid, nt, c1, Wq, vA, prev_c, kwh, nh, qW, sc, red, outv);
    for (int d = tid; d < EMBD; d += nt) c1[d] = dh[d] + outv[d];  // c0 = h0n + attn(c0,...)
    __syncthreads();
  }
  for (int d = tid; d < EMBD; d += nt) { h1[d] = dh[d]; h2[d] = dh[d]; c2[d] = c1[d]; }
  __syncthreads();

  // ---- decode ----
  for (int t = 0; t < STEPS; ++t) {
    int tok = gt_row[t];
    const float* src = (tok < VQ_) ? (query_emb + (size_t)clampi(tok, 0, VQ_ - 1) * EMBD)
                                   : (S + (size_t)clampi(tok - VQ_, 0, NCOLS - 1) * EMBD);
    for (int d = tid; d < EMBD; d += nt) xin[d] = src[d];
    __syncthreads();

    attn_ctx(tid, nt, dh, Wq + 1 * EMBD * EMBD, vA + 1 * EMBD, S,        KWs, NCOLS, qW, sc, red, xin + EMBD);
    attn_ctx(tid, nt, dh, Wq + 2 * EMBD * EMBD, vA + 2 * EMBD, keys_in,  KWi, nki,   qW, sc, red, xin + 2 * EMBD);
    if (nko > 0) {
      attn_ctx(tid, nt, dh, Wq + 3 * EMBD * EMBD, vA + 3 * EMBD, keys_out, KWo, nko, qW, sc, red, xin + 3 * EMBD);
    } else {
      for (int d = tid; d < EMBD; d += nt) xin[3 * EMBD + d] = 0.0f;
      __syncthreads();
    }

    // LSTM cell 1:  z = Wih0 * xin + Whh0 * h1 + b0
    for (int r = tid; r < 4 * DECSZ; r += nt) {
      const float* wi = Wih0 + (size_t)r * (4 * EMBD);
      const float* wh = Whh0 + (size_t)r * DECSZ;
      float acc = b0[r];
      for (int k = 0; k < 4 * EMBD; ++k) acc += wi[k] * xin[k];
      for (int k = 0; k < DECSZ; ++k)    acc += wh[k] * h1[k];
      z[r] = acc;
    }
    __syncthreads();
    for (int d = tid; d < DECSZ; d += nt) {
      float ig = sigf(z[d]), fg = sigf(z[DECSZ + d]);
      float gg = tanhf(z[2 * DECSZ + d]), og = sigf(z[3 * DECSZ + d]);
      float cn = fg * c1[d] + ig * gg;
      c1[d] = cn; h1[d] = og * tanhf(cn);
    }
    __syncthreads();

    // LSTM cell 2
    for (int r = tid; r < 4 * DECSZ; r += nt) {
      const float* wi = Wih1 + (size_t)r * DECSZ;
      const float* wh = Whh1 + (size_t)r * DECSZ;
      float acc = b1[r];
      for (int k = 0; k < DECSZ; ++k) acc += wi[k] * h1[k] + wh[k] * h2[k];
      z[r] = acc;
    }
    __syncthreads();
    for (int d = tid; d < DECSZ; d += nt) {
      float ig = sigf(z[d]), fg = sigf(z[DECSZ + d]);
      float gg = tanhf(z[2 * DECSZ + d]), og = sigf(z[3 * DECSZ + d]);
      float cn = fg * c2[d] + ig * gg;
      c2[d] = cn; h2[d] = og * tanhf(cn);
    }
    __syncthreads();

    // out = tanh([h2, ctx] @ Wt^T)
    for (int d = tid; d < EMBD; d += nt) {
      const float* w = Wt + (size_t)d * (4 * EMBD);
      float s = 0.0f;
      for (int k = 0; k < DECSZ; ++k)          s += w[k] * h2[k];
      for (int k = EMBD; k < 4 * EMBD; ++k)    s += w[k] * xin[k];
      out_vecs[(size_t)t * EMBD + d] = tanhf(s);
    }
    __syncthreads();
    for (int d = tid; d < EMBD; d += nt) dh[d] = h2[d];
    __syncthreads();
  }
}

// ===================================================================
// Launch
// ===================================================================
extern "C" void kernel_launch(void* const* d_in, const int* in_sizes, int n_in,
                              void* d_out, int out_size, void* d_ws, size_t ws_size,
                              hipStream_t stream) {
  (void)in_sizes; (void)n_in; (void)out_size; (void)ws_size;
  const float* utter_emb      = (const float*)d_in[0];
  const float* query_emb      = (const float*)d_in[1];
  const float* schema_tok_emb = (const float*)d_in[2];
  const float* enc_Wih        = (const float*)d_in[3];   // [3,2,600,300]
  const float* enc_Whh        = (const float*)d_in[4];   // [3,2,600,150]
  const float* enc_b          = (const float*)d_in[5];   // [3,2,600]
  const float* attn_Wq        = (const float*)d_in[6];   // [4,300,300]
  const float* attn_Wk        = (const float*)d_in[7];
  const float* attn_v         = (const float*)d_in[8];   // [4,300]
  const float* dec_Wih0       = (const float*)d_in[9];
  const float* dec_Whh0       = (const float*)d_in[10];
  const float* dec_b0         = (const float*)d_in[11];
  const float* dec_Wih1       = (const float*)d_in[12];
  const float* dec_Whh1       = (const float*)d_in[13];
  const float* dec_b1         = (const float*)d_in[14];
  const float* transform_W    = (const float*)d_in[15];
  const float* sql_W          = (const float*)d_in[16];
  const float* sql_b          = (const float*)d_in[17];
  const float* col_W          = (const float*)d_in[18];
  const int*   schema_tokens  = (const int*)d_in[19];
  const int*   input_seqs     = (const int*)d_in[20];
  const int*   output_gt      = (const int*)d_in[21];
  float* out = (float*)d_out;
  float* ws  = (float*)d_ws;

  // ---- workspace layout (floats) ----
  float* S        = ws;                    // 64*300
  float* x_embed  = S        + 19200;      // 128*300
  float* y1       = x_embed  + 38400;      // 128*300
  float* y2buf    = y1       + 38400;      // 128*300
  float* zin      = y2buf    + 38400;      // 384*1200
  float* prev_in  = zin      + 460800;     // 5*128*300
  float* prev_h   = prev_in  + 192000;     // 5*300
  float* prev_c   = prev_h   + 1500;       // 5*300
  float* prev_out = prev_c   + 1500;       // 5*64*300
  float* fhbuf    = prev_out + 96000;      // 300
  float* fcbuf    = fhbuf    + 300;        // 300
  float* attbuf   = fcbuf    + 300;        // 64*128
  float* colproj  = attbuf   + 8192;       // 63*300
  float* KWs      = colproj  + 18900;      // 64*300
  float* KWi      = KWs      + 19200;      // 640*300
  float* KWo      = KWi      + 192000;     // 320*300
  float* out_vecs = KWo      + 96000;      // 63*300
  float* coltok   = out_vecs + 18900;      // 384*300
  float* dummy    = coltok   + 115200;     // 600 scratch finals

  // ---- schema column BiLSTM ----
  k_coltok_embed<<<NCOLS * COLLEN, 128, 0, stream>>>(schema_tokens, schema_tok_emb, coltok);
  gemm(coltok, EMBD, enc_Wih, EMBD, 1, enc_b, 0.0f, zin, 1200, 0,
       NCOLS * COLLEN, 1200, EMBD, stream);
  k_schema_scan<<<NCOLS * 2, 192, 0, stream>>>(zin, enc_Whh, S);

  for (int i = 0; i < NTURNS; ++i) {
    float* x2 = (i < 5) ? (prev_in + (size_t)i * 38400) : y2buf;

    // mixed-vocab embedding + two stacked BiLSTMs (input-proj on WMMA)
    k_embed_input<<<INLEN, 128, 0, stream>>>(input_seqs + i * INLEN, utter_emb, S, x_embed);
    gemm(x_embed, EMBD, enc_Wih + 360000, EMBD, 1, enc_b + 1200, 0.0f, zin, 1200, 0,
         INLEN, 1200, EMBD, stream);
    k_lstm_scan<<<2, 640, 0, stream>>>(zin, INLEN, enc_Whh + 180000, y1, dummy, dummy + 300);
    gemm(y1, EMBD, enc_Wih + 720000, EMBD, 1, enc_b + 2400, 0.0f, zin, 1200, 0,
         INLEN, 1200, EMBD, stream);
    k_lstm_scan<<<2, 640, 0, stream>>>(zin, INLEN, enc_Whh + 360000, x2, fhbuf, fcbuf);
    if (i < 5)
      k_save_state<<<1, 320, 0, stream>>>(fhbuf, fcbuf, prev_h + i * EMBD, prev_c + i * EMBD);

    // schema cross-attention update: S += softmax(S X^T) X
    gemm(S, EMBD, x2, EMBD, 1, nullptr, 0.0f, attbuf, INLEN, 0, NCOLS, INLEN, EMBD, stream);
    k_softmax_rows<<<NCOLS, 128, 0, stream>>>(attbuf, INLEN);
    gemm(attbuf, INLEN, x2, EMBD, 0, nullptr, 1.0f, S, EMBD, 0, NCOLS, EMBD, INLEN, stream);

    // per-turn attention key projections
    int nki = imin2(i + 1, 5) * INLEN;
    int nko = imin2(i, 5) * GENLEN;
    int nh  = imin2(i, 5);
    gemm(S, EMBD, attn_Wk + 1 * 90000, EMBD, 0, nullptr, 0.0f, KWs, EMBD, 0,
         NCOLS, EMBD, EMBD, stream);
    gemm(prev_in, EMBD, attn_Wk + 2 * 90000, EMBD, 0, nullptr, 0.0f, KWi, EMBD, 0,
         nki, EMBD, EMBD, stream);
    if (nko > 0)
      gemm(prev_out, EMBD, attn_Wk + 3 * 90000, EMBD, 0, nullptr, 0.0f, KWo, EMBD, 0,
           nko, EMBD, EMBD, stream);

    // sequential decoder scan (collects 63 `out` vectors)
    k_decoder<<<1, 1024, 0, stream>>>(fhbuf, fcbuf, prev_h, prev_c, nh, S, KWs,
                                      prev_in, KWi, nki, prev_out, KWo, nko,
                                      attn_Wq, attn_Wk, attn_v,
                                      dec_Wih0, dec_Whh0, dec_b0,
                                      dec_Wih1, dec_Whh1, dec_b1,
                                      transform_W, output_gt + i * GENLEN, query_emb,
                                      out_vecs);

    // batched score projections on WMMA: [63x300] x vocab / columns
    float* od = out + (size_t)i * STEPS * NOUT;
    gemm(out_vecs, EMBD, sql_W, EMBD, 1, sql_b, 0.0f, od, NOUT, 0,
         STEPS, VQ_, EMBD, stream);
    gemm(out_vecs, EMBD, col_W, EMBD, 1, nullptr, 0.0f, colproj, EMBD, 0,
         STEPS, EMBD, EMBD, stream);
    gemm(colproj, EMBD, S, EMBD, 1, nullptr, 0.0f, od, NOUT, VQ_,
         STEPS, NCOLS, EMBD, stream);

    k_logsoftmax_rows<<<STEPS, 256, 0, stream>>>(od, NOUT);
    if (i < 5)
      k_amemb<<<STEPS, 256, 0, stream>>>(od, query_emb, S, prev_out + (size_t)i * 19200);
  }
}
